// BlockSparseDTM_11948599017534
// MI455X (gfx1250) — compile-verified, matrix operand order checked by reading, not run on hardware
//
#include <hip/hip_runtime.h>
#include <hip/hip_bf16.h>
#include <stdint.h>

typedef _Float16 f16;
typedef _Float16 v16h __attribute__((ext_vector_type(16)));
typedef _Float16 v8h  __attribute__((ext_vector_type(8)));
typedef float    v8f  __attribute__((ext_vector_type(8)));

#define DD        192                 // p-bits per block
#define NROWS     16384               // BATCH * N = 256*64
#define HALF_ELEMS 1572864u           // NROWS*DD/2 (threefry counter split)
#define XSTRIDE   200                 // padded LDS row stride (halves): bank-conflict-free
#define T_STEPS   100
#define SWEEPS    6
#define WAVES_PER_BLOCK 8
#define NBLOCKS   64                  // 64 blocks * 8 waves = 512 tile pairs = 1024 tiles

// ---------------- Threefry-2x32 (exact JAX/Random123 schedule) ----------------
__device__ __forceinline__ uint32_t rotl32(uint32_t x, int r) {
  return (x << r) | (x >> (32 - r));
}
__device__ __forceinline__ void tf_round(uint32_t& x0, uint32_t& x1, int r) {
  x0 += x1; x1 = rotl32(x1, r); x1 ^= x0;
}
__device__ __forceinline__ void threefry2x32(uint32_t k0, uint32_t k1,
                                             uint32_t x0, uint32_t x1,
                                             uint32_t& o0, uint32_t& o1) {
  uint32_t k2 = 0x1BD11BDAu ^ k0 ^ k1;
  x0 += k0; x1 += k1;
  tf_round(x0,x1,13); tf_round(x0,x1,15); tf_round(x0,x1,26); tf_round(x0,x1, 6);
  x0 += k1; x1 += k2 + 1u;
  tf_round(x0,x1,17); tf_round(x0,x1,29); tf_round(x0,x1,16); tf_round(x0,x1,24);
  x0 += k2; x1 += k0 + 2u;
  tf_round(x0,x1,13); tf_round(x0,x1,15); tf_round(x0,x1,26); tf_round(x0,x1, 6);
  x0 += k0; x1 += k1 + 3u;
  tf_round(x0,x1,17); tf_round(x0,x1,29); tf_round(x0,x1,16); tf_round(x0,x1,24);
  x0 += k1; x1 += k2 + 4u;
  tf_round(x0,x1,13); tf_round(x0,x1,15); tf_round(x0,x1,26); tf_round(x0,x1, 6);
  x0 += k2; x1 += k0 + 5u;
  o0 = x0; o1 = x1;
}
__device__ __forceinline__ float bits_to_uniform(uint32_t b) {
  return __uint_as_float((b >> 9) | 0x3f800000u) - 1.0f;
}

// ---------------- K1: M[i,j] = sum_o Wl*mask_l + 0.3*sum_o Ws*mask_s ----------------
__global__ void kA_reduceM(const float* __restrict__ Wl, const float* __restrict__ ml,
                           const float* __restrict__ Ws, const float* __restrict__ ms,
                           float* __restrict__ M) {
  int idx = blockIdx.x * 256 + threadIdx.x;
  if (idx >= DD * DD) return;
  float a = 0.f, b = 0.f;
  for (int o = 0; o < DD; ++o) {
    int p = o * DD * DD + idx;
    a += Wl[p] * ml[p];
    b += Ws[p] * ms[p];
  }
  M[idx] = a + 0.3f * b;
}

// ---------------- K2: S = (f16)(M + M^T), dM = diag(M) ----------------
__global__ void kB_makeS(const float* __restrict__ M, f16* __restrict__ Sh,
                         float* __restrict__ dM) {
  int idx = blockIdx.x * 256 + threadIdx.x;
  if (idx >= DD * DD) return;
  int i = idx / DD, j = idx % DD;
  Sh[idx] = (f16)(M[i * DD + j] + M[j * DD + i]);
  if (i == j) dM[i] = M[idx];
}

// ---------------- K3: c[t,d] = bias + time_proj_b + time_emb[t] . time_proj_w[d] ----------------
__global__ void kC_cAll(const float* __restrict__ temb, const float* __restrict__ tpw,
                        const float* __restrict__ tpb, const float* __restrict__ bias,
                        float* __restrict__ c_all) {
  int idx = blockIdx.x * 256 + threadIdx.x;
  if (idx >= T_STEPS * DD) return;
  int t = idx / DD, d = idx % DD;
  float a = bias[d] + tpb[d];
  for (int e = 0; e < 32; ++e) a += temb[t * 32 + e] * tpw[d * 32 + e];
  c_all[idx] = a;
}

// ---------------- K4: fused 600-sweep Gibbs sampler, WMMA mat-vec, per-wave tiles ----------------
__global__ __launch_bounds__(256) void kMain(const int* __restrict__ x_init,
                                             const f16* __restrict__ Sg,
                                             const float* __restrict__ dMg,
                                             const float* __restrict__ c_all,
                                             float* __restrict__ out) {
  extern __shared__ __attribute__((aligned(128))) char smem_raw[];
  f16* Slds = (f16*)smem_raw;                    // DD * XSTRIDE halves
  f16* Xall = Slds + DD * XSTRIDE;               // 8 waves * 2 tiles * 16 * XSTRIDE

  const int tid  = threadIdx.x;
  const int lane = tid & 31;
  const int wave = tid >> 5;
  const int ln   = lane & 15;   // A: row M ; B: column N ; D: column N
  const int hi   = lane >> 4;   // half-wave selector (K / row-group offsets)

  // -- stage S into LDS (padded stride), all threads --
  const uint32_t* Sg32 = (const uint32_t*)Sg;
  for (int i = tid; i < DD * (DD / 2); i += 256) {
    int r = i / (DD / 2), c = i % (DD / 2);
    *(uint32_t*)(Slds + r * XSTRIDE + c * 2) = Sg32[i];
  }

  // -- per-wave pair of 16-row tiles: rows [baseA..baseA+16) and +8192 (threefry pairing) --
  const int pair  = blockIdx.x * WAVES_PER_BLOCK + wave;   // 0..511
  const int baseA = pair * 16;
  const int baseB = baseA + 8192;
  f16* Xw = Xall + wave * (2 * 16 * XSTRIDE);

  // -- blockify x_init into per-wave LDS tiles (binary f16) --
  for (int e = lane; e < 2 * 16 * DD; e += 32) {
    int tile = e / (16 * DD);
    int rem  = e % (16 * DD);
    int r = rem / DD, d = rem % DD;
    int grow = (tile ? baseB : baseA) + r;
    int b = grow >> 6, nblk = grow & 63;
    int by = nblk >> 3, bx = nblk & 7;
    int pix = d / 3, ch = d % 3;
    int H = by * 8 + (pix >> 3), W = bx * 8 + (pix & 7);
    int src = ((b * 3 + ch) * 64 + H) * 64 + W;
    Xw[tile * (16 * XSTRIDE) + r * XSTRIDE + d] = (f16)(x_init[src] > 0 ? 1.0f : 0.0f);
  }
  __syncthreads();

  for (int tt = 0; tt < T_STEPS; ++tt) {
    const int t = T_STEPS - 1 - tt;            // reversed diffusion steps
    for (int s = 0; s < SWEEPS; ++s) {
      uint32_t k0, k1;                          // fold_in(key(42), t*6+s)
      threefry2x32(0u, 42u, 0u, (uint32_t)(t * SWEEPS + s), k0, k1);

      // snapshot A fragments for the whole sweep (synchronous Gibbs)
      v16h A[2][6];
#pragma unroll
      for (int tile = 0; tile < 2; ++tile) {
#pragma unroll
        for (int kc = 0; kc < 6; ++kc) {
          const f16* p = Xw + tile * (16 * XSTRIDE) + ln * XSTRIDE + kc * 32 + hi * 8;
          v8h lo = *(const v8h*)(p);
          v8h hh = *(const v8h*)(p + 16);
          A[tile][kc] = __builtin_shufflevector(lo, hh, 0,1,2,3,4,5,6,7,8,9,10,11,12,13,14,15);
        }
      }

      for (int ct = 0; ct < 12; ++ct) {         // 12 output column tiles of 16
        const int kglob = ct * 16 + ln;

        // load ALL 6 B fragments up front (12 back-to-back ds_load_b128, single wait)
        v16h B[6];
#pragma unroll
        for (int kc = 0; kc < 6; ++kc) {
          const f16* p = Slds + kglob * XSTRIDE + kc * 32 + hi * 8;
          v8h lo = *(const v8h*)(p);
          v8h hh = *(const v8h*)(p + 16);
          B[kc] = __builtin_shufflevector(lo, hh, 0,1,2,3,4,5,6,7,8,9,10,11,12,13,14,15);
        }

        // hoist old-x reads: hidden under the WMMA burst
        float xAold[8], xBold[8];
#pragma unroll
        for (int v = 0; v < 8; ++v) {
          const int rl = v + 8 * hi;
          xAold[v] = (float)Xw[rl * XSTRIDE + kglob];
          xBold[v] = (float)Xw[(16 + rl) * XSTRIDE + kglob];
        }

        v8f accA = {}; v8f accB = {};
#pragma unroll
        for (int kc = 0; kc < 6; ++kc) {
          accA = __builtin_amdgcn_wmma_f32_16x16x32_f16(false, A[0][kc], false, B[kc],
                                                        (short)0, accA, false, false);
          accB = __builtin_amdgcn_wmma_f32_16x16x32_f16(false, A[1][kc], false, B[kc],
                                                        (short)0, accB, false, false);
        }

        const float dm = dMg[kglob];
        const float cc = c_all[t * DD + kglob];

#pragma unroll
        for (int v = 0; v < 8; ++v) {
          const int rl = v + 8 * hi;            // D-layout row within tile
          uint32_t j = (uint32_t)((baseA + rl) * DD + kglob);
          uint32_t b0, b1;
          threefry2x32(k0, k1, j, j + HALF_ELEMS, b0, b1);   // one call -> both tiles
          float uA = bits_to_uniform(b0);
          float uB = bits_to_uniform(b1);
          // -dE = field - 2*dM*x + dM + c ; p = sigmoid(-dE)
          float argA = accA[v] + dm * (1.0f - 2.0f * xAold[v]) + cc;
          float argB = accB[v] + dm * (1.0f - 2.0f * xBold[v]) + cc;
          float pa = 1.0f / (1.0f + __expf(-argA));
          float pb = 1.0f / (1.0f + __expf(-argB));
          Xw[rl * XSTRIDE + kglob]        = (f16)(uA < pa ? 1.0f : 0.0f);
          Xw[(16 + rl) * XSTRIDE + kglob] = (f16)(uB < pb ? 1.0f : 0.0f);
        }
      }
    }
  }

  // -- unblockify to [B,C,H,W] float32 --
  for (int e = lane; e < 2 * 16 * DD; e += 32) {
    int tile = e / (16 * DD);
    int rem  = e % (16 * DD);
    int r = rem / DD, d = rem % DD;
    int grow = (tile ? baseB : baseA) + r;
    int b = grow >> 6, nblk = grow & 63;
    int by = nblk >> 3, bx = nblk & 7;
    int pix = d / 3, ch = d % 3;
    int H = by * 8 + (pix >> 3), W = bx * 8 + (pix & 7);
    out[((b * 3 + ch) * 64 + H) * 64 + W] =
        (float)Xw[tile * (16 * XSTRIDE) + r * XSTRIDE + d];
  }
}

extern "C" void kernel_launch(void* const* d_in, const int* in_sizes, int n_in,
                              void* d_out, int out_size, void* d_ws, size_t ws_size,
                              hipStream_t stream) {
  (void)in_sizes; (void)n_in; (void)out_size; (void)ws_size;
  const int*   x_init = (const int*)  d_in[0];
  const float* temb   = (const float*)d_in[1];
  const float* tpw    = (const float*)d_in[2];
  const float* tpb    = (const float*)d_in[3];
  const float* Wl     = (const float*)d_in[4];
  const float* ml     = (const float*)d_in[5];
  const float* Ws     = (const float*)d_in[6];
  const float* ms     = (const float*)d_in[7];
  const float* bias   = (const float*)d_in[8];

  char* ws = (char*)d_ws;
  float* M     = (float*)(ws);                               // 147456 B
  f16*   Sh    = (f16*)  (ws + 147456);                      //  73728 B
  float* dM    = (float*)(ws + 147456 + 73728);              //    768 B
  float* c_all = (float*)(ws + 147456 + 73728 + 768);        //  76800 B

  kA_reduceM<<<(DD * DD + 255) / 256, 256, 0, stream>>>(Wl, ml, Ws, ms, M);
  kB_makeS  <<<(DD * DD + 255) / 256, 256, 0, stream>>>(M, Sh, dM);
  kC_cAll   <<<(T_STEPS * DD + 255) / 256, 256, 0, stream>>>(temb, tpw, tpb, bias, c_all);

  size_t lds_bytes = (size_t)(DD * XSTRIDE +
                              WAVES_PER_BLOCK * 2 * 16 * XSTRIDE) * sizeof(f16); // 179200 B
  kMain<<<NBLOCKS, WAVES_PER_BLOCK * 32, lds_bytes, stream>>>(x_init, Sh, dM, c_all,
                                                              (float*)d_out);
}
